// VAE_m2_27693949125146
// MI455X (gfx1250) — matrix operand order repaired and checked by reading, not compiled, not run on hardware
//
#include <hip/hip_runtime.h>
#include <hip/hip_bf16.h>

typedef __attribute__((ext_vector_type(16))) _Float16 v16h;
typedef __attribute__((ext_vector_type(8)))  _Float16 v8h;
typedef __attribute__((ext_vector_type(8)))  float    v8f;

__device__ __forceinline__ float sigmoidf_(float x) {
    return 1.0f / (1.0f + __expf(-x));
}

// ---------------------------------------------------------------------------
// Prep kernels: f32 -> f16 with K padded to multiple of 32; weights transposed
// ---------------------------------------------------------------------------

// src (R x C) f32 row-major -> dst (R x ldd) f16, zero pad cols [C, ldd)
__global__ void k_cvt_pad(const float* __restrict__ src, _Float16* __restrict__ dst,
                          int R, int C, int ldd) {
    int i = blockIdx.x * blockDim.x + threadIdx.x;
    if (i >= R * ldd) return;
    int r = i / ldd, c = i - r * ldd;
    dst[i] = (c < C) ? (_Float16)src[(size_t)r * C + c] : (_Float16)0.0f;
}

// src (K x N) f32 row-major -> dst (N x Kp) f16: dst[n*Kp+k] = src[k*N+n], zero pad k >= K
__global__ void k_tr_cvt(const float* __restrict__ src, _Float16* __restrict__ dst,
                         int K, int N, int Kp) {
    int i = blockIdx.x * blockDim.x + threadIdx.x;
    if (i >= N * Kp) return;
    int n = i / Kp, k = i - n * Kp;
    dst[i] = (k < K) ? (_Float16)src[(size_t)k * N + n] : (_Float16)0.0f;
}

// ---------------------------------------------------------------------------
// WMMA GEMM: C[M x N] = act(A[M x Kp] @ Bt[N x Kp]^T + bias)
//   block = 256 threads = 8 waves (4 wave-rows x 2 wave-cols) -> 128x128 tile
//   wave  = 32x64 strip = 2x4 tiles of v_wmma_f32_16x16x32_f16
//   Branch-free, ping-pong double-buffered K-loop (2 steps/iter, no rotation
//   copies); B columns clamped so EXEC stays all-1s; stores masked in epilogue.
// ---------------------------------------------------------------------------
template<bool ACT, bool OUT16>
__global__ __launch_bounds__(256) void k_gemm_wmma(
    const _Float16* __restrict__ A,    // M x Kp, row-major (Kp % 32 == 0)
    const _Float16* __restrict__ Bt,   // N x Kp, row-major (= B transposed)
    const float*    __restrict__ bias, // N
    _Float16* __restrict__ C16, float* __restrict__ C32,
    int N, int Kp)
{
    const int lane = threadIdx.x & 31;
    const int wave = threadIdx.x >> 5;
    const int wr   = wave & 3;     // 4 wave-rows  -> 32-row strips
    const int wc   = wave >> 2;    // 2 wave-cols  -> 64-col strips
    const int lm   = lane & 15;
    const int hs   = lane >> 4;    // half-wave select

    const int mBase = blockIdx.y * 128 + wr * 32;
    const int nBase = blockIdx.x * 128 + wc * 64;

    // Loop-invariant per-lane base pointers.
    // A frag: lane holds row (m0+lm), 16 contiguous K starting at hs*16
    const _Float16* aPtr[2];
    #pragma unroll
    for (int mt = 0; mt < 2; ++mt)
        aPtr[mt] = A + (size_t)(mBase + mt * 16 + lm) * Kp + hs * 16;

    // B frag: lane holds column (n0+lm), 16 contiguous K starting at hs*16.
    // Clamp out-of-range columns to N-1: always-valid loads, no divergence.
    const _Float16* bPtr[4];
    #pragma unroll
    for (int nt = 0; nt < 4; ++nt) {
        int n = nBase + nt * 16 + lm;
        if (n > N - 1) n = N - 1;
        bPtr[nt] = Bt + (size_t)n * Kp + hs * 16;
    }

    v8f acc[2][4];
    #pragma unroll
    for (int mt = 0; mt < 2; ++mt)
        #pragma unroll
        for (int nt = 0; nt < 4; ++nt) {
            v8f z = {};
            acc[mt][nt] = z;
        }

    #define LOAD_FRAGS(aDst, bDst, k0)                                          \
        do {                                                                    \
            _Pragma("unroll")                                                   \
            for (int mt = 0; mt < 2; ++mt) {                                    \
                v8h lo = *(const v8h*)(aPtr[mt] + (k0));                        \
                v8h hi = *(const v8h*)(aPtr[mt] + (k0) + 8);                    \
                aDst[mt] = __builtin_shufflevector(lo, hi,                      \
                    0, 1, 2, 3, 4, 5, 6, 7, 8, 9, 10, 11, 12, 13, 14, 15);      \
            }                                                                   \
            _Pragma("unroll")                                                   \
            for (int nt = 0; nt < 4; ++nt)                                      \
                bDst[nt] = *(const v16h*)(bPtr[nt] + (k0));                     \
        } while (0)

    #define MMA_FRAGS(aSrc, bSrc)                                               \
        do {                                                                    \
            _Pragma("unroll")                                                   \
            for (int mt = 0; mt < 2; ++mt)                                      \
                _Pragma("unroll")                                               \
                for (int nt = 0; nt < 4; ++nt)                                  \
                    acc[mt][nt] = __builtin_amdgcn_wmma_f32_16x16x32_f16(       \
                        false, aSrc[mt], false, bSrc[nt], (short)0,             \
                        acc[mt][nt], false, false);                             \
        } while (0)

    // Ping-pong software pipeline: 2 K-steps per iteration, no buffer copies.
    const int nsteps = Kp >> 5;    // >= 1; Kp % 32 == 0
    v16h a0[2], b0[4], a1[2], b1[4];
    LOAD_FRAGS(a0, b0, 0);
    int k = 32;
    int i = 1;
    for (; i + 1 < nsteps; i += 2) {
        LOAD_FRAGS(a1, b1, k);  k += 32;
        MMA_FRAGS(a0, b0);
        LOAD_FRAGS(a0, b0, k);  k += 32;
        MMA_FRAGS(a1, b1);
    }
    if (i < nsteps) {              // one trailing step (even nsteps)
        LOAD_FRAGS(a1, b1, k);
        MMA_FRAGS(a0, b0);
        MMA_FRAGS(a1, b1);
    } else {
        MMA_FRAGS(a0, b0);
    }

    #undef LOAD_FRAGS
    #undef MMA_FRAGS

    // Epilogue: lane owns column n = n0+lm, rows m0 + hs*8 + r
    #pragma unroll
    for (int mt = 0; mt < 2; ++mt) {
        #pragma unroll
        for (int nt = 0; nt < 4; ++nt) {
            int n = nBase + nt * 16 + lm;
            if (n >= N) continue;
            float bv = bias[n];
            int m0 = mBase + mt * 16 + hs * 8;
            #pragma unroll
            for (int r = 0; r < 8; ++r) {
                float v = acc[mt][nt][r] + bv;
                if (ACT) v = sigmoidf_(v);
                size_t off = (size_t)(m0 + r) * N + n;
                if (OUT16) C16[off] = (_Float16)v;
                else       C32[off] = v;
            }
        }
    }
}

// ---------------------------------------------------------------------------
// Reparameterization: z[b,l,:] = eps[l,:] * sd[b,:] + mean[b,:]   (f16 out)
// enc layout: (4096 x 128), mean = [:, :64], sd = [:, 64:]
// ---------------------------------------------------------------------------
__global__ void k_reparam(const float* __restrict__ enc, const float* __restrict__ eps,
                          _Float16* __restrict__ zt) {
    int i = blockIdx.x * blockDim.x + threadIdx.x;   // 65536 * 64
    if (i >= 65536 * 64) return;
    int z   = i & 63;
    int row = i >> 6;       // 0..65535 = b*16 + l
    int b   = row >> 4;
    int l   = row & 15;
    float mean = enc[b * 128 + z];
    float sd   = enc[b * 128 + 64 + z];
    zt[i] = (_Float16)(eps[l * 64 + z] * sd + mean);
}

// Scatter mean_b / sd_b into d_out after x_rec
__global__ void k_meansd(const float* __restrict__ enc, float* __restrict__ out) {
    int i = blockIdx.x * blockDim.x + threadIdx.x;   // 4096 * 128
    if (i >= 4096 * 128) return;
    int b = i >> 7, c = i & 127;
    const size_t XREC = (size_t)65536 * 784;
    float v = enc[i];
    if (c < 64) out[XREC + (size_t)b * 64 + c] = v;
    else        out[XREC + (size_t)4096 * 64 + (size_t)b * 64 + (c - 64)] = v;
}

// ---------------------------------------------------------------------------
// Launcher
// ---------------------------------------------------------------------------
extern "C" void kernel_launch(void* const* d_in, const int* in_sizes, int n_in,
                              void* d_out, int out_size, void* d_ws, size_t ws_size,
                              hipStream_t stream) {
    (void)in_sizes; (void)n_in; (void)out_size; (void)ws_size;

    const float* x   = (const float*)d_in[0];
    const float* eps = (const float*)d_in[1];
    const float* ew1 = (const float*)d_in[2];
    const float* eb1 = (const float*)d_in[3];
    const float* ew2 = (const float*)d_in[4];
    const float* eb2 = (const float*)d_in[5];
    const float* ew3 = (const float*)d_in[6];
    const float* eb3 = (const float*)d_in[7];
    const float* dw1 = (const float*)d_in[8];
    const float* db1 = (const float*)d_in[9];
    const float* dw2 = (const float*)d_in[10];
    const float* db2 = (const float*)d_in[11];
    const float* dw3 = (const float*)d_in[12];
    const float* db3 = (const float*)d_in[13];
    float* out = (float*)d_out;

    // Workspace carve-out
    char* w = (char*)d_ws;
    size_t off = 0;
    auto carve = [&](size_t bytes) {
        void* p = w + off;
        off += (bytes + 255) & ~(size_t)255;
        return p;
    };
    _Float16* xh   = (_Float16*)carve((size_t)4096  * 800  * 2); // x padded, K 784->800
    _Float16* ew1t = (_Float16*)carve((size_t)1024  * 800  * 2);
    _Float16* ew2t = (_Float16*)carve((size_t)512   * 1024 * 2);
    _Float16* ew3t = (_Float16*)carve((size_t)128   * 512  * 2);
    _Float16* dw1t = (_Float16*)carve((size_t)512   * 64   * 2);
    _Float16* dw2t = (_Float16*)carve((size_t)1024  * 512  * 2);
    _Float16* dw3t = (_Float16*)carve((size_t)784   * 1024 * 2);
    float*    enc  = (float*)   carve((size_t)4096  * 128  * 4);
    _Float16* zt   = (_Float16*)carve((size_t)65536 * 64   * 2);
    _Float16* big1 = (_Float16*)carve((size_t)65536 * 512  * 2); // h1 then g1
    _Float16* big2 = (_Float16*)carve((size_t)65536 * 1024 * 2); // h2 then g2

    auto blocks = [](long n) { return dim3((unsigned)((n + 255) / 256)); };

    // --- precision conversion / weight transposition ---
    k_cvt_pad<<<blocks((long)4096 * 800), 256, 0, stream>>>(x, xh, 4096, 784, 800);
    k_tr_cvt <<<blocks((long)1024 * 800), 256, 0, stream>>>(ew1, ew1t, 784, 1024, 800);
    k_tr_cvt <<<blocks((long)512 * 1024), 256, 0, stream>>>(ew2, ew2t, 1024, 512, 1024);
    k_tr_cvt <<<blocks((long)128 * 512),  256, 0, stream>>>(ew3, ew3t, 512, 128, 512);
    k_tr_cvt <<<blocks((long)512 * 64),   256, 0, stream>>>(dw1, dw1t, 64, 512, 64);
    k_tr_cvt <<<blocks((long)1024 * 512), 256, 0, stream>>>(dw2, dw2t, 512, 1024, 512);
    k_tr_cvt <<<blocks((long)784 * 1024), 256, 0, stream>>>(dw3, dw3t, 1024, 784, 1024);

    // --- encoder ---
    k_gemm_wmma<true,  true ><<<dim3(8, 32),  256, 0, stream>>>(xh,   ew1t, eb1, big1, nullptr, 1024, 800);
    k_gemm_wmma<true,  true ><<<dim3(4, 32),  256, 0, stream>>>(big1, ew2t, eb2, big2, nullptr, 512,  1024);
    k_gemm_wmma<false, false><<<dim3(1, 32),  256, 0, stream>>>(big2, ew3t, eb3, nullptr, enc,  128,  512);

    // --- reparameterization + mean/sd outputs ---
    k_reparam<<<blocks((long)65536 * 64), 256, 0, stream>>>(enc, eps, zt);
    k_meansd <<<blocks((long)4096 * 128), 256, 0, stream>>>(enc, out);

    // --- decoder ---
    k_gemm_wmma<true, true ><<<dim3(4, 512), 256, 0, stream>>>(zt,   dw1t, db1, big1, nullptr, 512,  64);
    k_gemm_wmma<true, true ><<<dim3(8, 512), 256, 0, stream>>>(big1, dw2t, db2, big2, nullptr, 1024, 512);
    k_gemm_wmma<true, false><<<dim3(7, 512), 256, 0, stream>>>(big2, dw3t, db3, nullptr, out,  784,  1024);
}
